// Conttime_23330262352257
// MI455X (gfx1250) — matrix operand order repaired and checked by reading, compiled, not run on hardware
//
#include <hip/hip_runtime.h>
#include <hip/hip_bf16.h>

typedef __attribute__((ext_vector_type(16))) __bf16        v16bf;
typedef __attribute__((ext_vector_type(8)))  float         v8f;
typedef __attribute__((ext_vector_type(4)))  unsigned int  v4u;

#define T_STEPS 1024
#define B_TOT   512
#define HID     128
#define GDIM    896      // 7*HID
#define KDIM    384      // 3*HID
#define KT      12       // 384/32 k-tiles
#define NT      56       // 896/16 n-tiles
#define BT      16       // batch rows per workgroup

__device__ __forceinline__ unsigned short f2bf(float f) {
    unsigned int u = __builtin_bit_cast(unsigned int, f);
    unsigned int lsb = (u >> 16) & 1u;
    u += 0x7fffu + lsb;                       // round-to-nearest-even
    return (unsigned short)(u >> 16);
}
__device__ __forceinline__ float sigm(float x)      { return 1.0f / (1.0f + __expf(-x)); }
__device__ __forceinline__ float softplus1(float x) { return fmaxf(x, 0.0f) + __logf(1.0f + __expf(-fabsf(x))); }

// ---------------------------------------------------------------------------
// Kernel 0: reorder W (384x896 f32, row-major) into bf16 WMMA B-fragments.
// Fragment (kt,nt) is 512 bf16, stored lane-major: element (lane, e) holds
// W[kt*32 + k][nt*16 + lane%16], k = 8*(lane/16) + e for e<8, else 16 + 8*(lane/16) + e-8.
// ---------------------------------------------------------------------------
__global__ void wconv_kernel(const float* __restrict__ W,
                             unsigned short* __restrict__ Wbf) {
    int tid = blockIdx.x * blockDim.x + threadIdx.x;
    if (tid >= KDIM * GDIM) return;
    int e    = tid & 15;
    int lane = (tid >> 4) & 31;
    int frag = tid >> 9;                 // kt*NT + nt
    int kt   = frag / NT;
    int nt   = frag - kt * NT;
    int k = kt * 32 + ((e & 8) << 1) + ((lane >> 4) << 3) + (e & 7);
    int n = nt * 16 + (lane & 15);
    Wbf[tid] = f2bf(W[k * GDIM + n]);
}

// ---------------------------------------------------------------------------
// Kernel 1: the CTLSTM scan. 32 workgroups x 16 batch rows, 8 waves each.
// Wave w owns hidden columns [16w,16w+16); keeps c / cbar resident in the
// 16x16 f32 accumulator layout for all 1024 steps.
// ---------------------------------------------------------------------------
__global__ __launch_bounds__(256) void ctlstm_kernel(
    const int*   __restrict__ items,
    const float* __restrict__ time_interval,
    const int*   __restrict__ durs,
    const float* __restrict__ item_emb,
    const float* __restrict__ dur_emb,
    const float* __restrict__ bias,
    const unsigned short* __restrict__ Wbf,
    float* __restrict__ h_out)
{
    __shared__ unsigned short Alds[BT * KDIM];   // 16 x 384 bf16 = [ie | de | h]
    __shared__ float dtl[BT];

    const int tid   = threadIdx.x;
    const int lane  = tid & 31;
    const int wv    = tid >> 5;           // wave 0..7
    const int bbase = blockIdx.x * BT;

    // zero A (h region must start at 0; h0 = c0 = cbar0 = 0)
    for (int i = tid; i < BT * KDIM; i += 256) Alds[i] = 0;

    // per-lane constants (C/D layout: M = r + 8*(lane>>4), N = lane&15)
    const int nloc  = lane & 15;
    const int mhalf = lane >> 4;          // 0 or 1
    const int hc    = wv * 16;            // hidden-column base of this wave
    const int arow  = lane & 15;          // A-frag: M = lane%16
    const int kb    = mhalf * 8;          // A-frag per-lane K base

    float biasr[7];
    #pragma unroll
    for (int g = 0; g < 7; ++g) biasr[g] = bias[g * HID + hc + nloc];

    float c[8], cbv[8];
    #pragma unroll
    for (int r = 0; r < 8; ++r) { c[r] = 0.0f; cbv[r] = 0.0f; }

    // gather mapping: 256 threads fill 16 rows x 256 embedding cols (16 each)
    const int grow = tid >> 4;            // row 0..15
    const int gcb  = (tid & 15) << 4;     // col block 0..240

    for (int t = 0; t < T_STEPS; ++t) {
        // ---- phase 1: stage embeddings (cols 0..255) + dt ----
        {
            const int bg = bbase + grow;
            const float* src;
            if (gcb < HID) {
                int it = items[bg * T_STEPS + t];
                src = item_emb + (size_t)it * HID + gcb;
            } else {
                int du = durs[bg * T_STEPS + t];
                src = dur_emb + (size_t)du * HID + (gcb - HID);
            }
            float4 f0 = ((const float4*)src)[0];
            float4 f1 = ((const float4*)src)[1];
            float4 f2 = ((const float4*)src)[2];
            float4 f3 = ((const float4*)src)[3];
            v4u p0, p1;
            p0.x = (unsigned)f2bf(f0.x) | ((unsigned)f2bf(f0.y) << 16);
            p0.y = (unsigned)f2bf(f0.z) | ((unsigned)f2bf(f0.w) << 16);
            p0.z = (unsigned)f2bf(f1.x) | ((unsigned)f2bf(f1.y) << 16);
            p0.w = (unsigned)f2bf(f1.z) | ((unsigned)f2bf(f1.w) << 16);
            p1.x = (unsigned)f2bf(f2.x) | ((unsigned)f2bf(f2.y) << 16);
            p1.y = (unsigned)f2bf(f2.z) | ((unsigned)f2bf(f2.w) << 16);
            p1.z = (unsigned)f2bf(f3.x) | ((unsigned)f2bf(f3.y) << 16);
            p1.w = (unsigned)f2bf(f3.z) | ((unsigned)f2bf(f3.w) << 16);
            v4u* dst = (v4u*)&Alds[grow * KDIM + gcb];
            dst[0] = p0;
            dst[1] = p1;
        }
        if (tid < BT) dtl[tid] = time_interval[(bbase + tid) * T_STEPS + t];
        __syncthreads();

        // ---- phase 2: g = [ie|de|h] @ W + b via WMMA (reads all of Alds) ----
        v8f acc[7];
        #pragma unroll
        for (int g = 0; g < 7; ++g) {
            v8f a0;
            #pragma unroll
            for (int r = 0; r < 8; ++r) a0[r] = biasr[g];
            acc[g] = a0;
        }
        for (int kt = 0; kt < KT; ++kt) {
            // A fragment (16-bit 16x32 layout): two contiguous 8-elem runs
            union { v16bf v; v4u q[2]; } af;
            const unsigned short* ap = &Alds[arow * KDIM + kt * 32 + kb];
            af.q[0] = *(const v4u*)ap;          // K = kb .. kb+7
            af.q[1] = *(const v4u*)(ap + 16);   // K = kb+16 .. kb+23
            #pragma unroll
            for (int g = 0; g < 7; ++g) {
                union { v16bf v; v4u q[2]; } bfr;
                const unsigned short* bp =
                    Wbf + (((size_t)(kt * NT + g * 8 + wv)) << 9) + lane * 16;
                bfr.q[0] = *(const v4u*)bp;
                bfr.q[1] = *(const v4u*)(bp + 8);
                acc[g] = __builtin_amdgcn_wmma_f32_16x16x32_bf16(
                    false, af.v, false, bfr.v, (short)0, acc[g], false, false);
            }
        }
        __syncthreads();   // all waves done reading Alds before h is rewritten

        // ---- phase 3: gate math, state update, h writeback ----
        float* hrow = h_out + (size_t)t * (B_TOT * HID)
                            + (size_t)(bbase + mhalf * 8) * HID + hc + nloc;
        #pragma unroll
        for (int r = 0; r < 8; ++r) {
            float gi  = acc[0][r], gf  = acc[1][r], go = acc[2][r], gz = acc[3][r];
            float gib = acc[4][r], gfb = acc[5][r], gd = acc[6][r];
            float iv  = sigm(gi);
            float fv  = sigm(gf);
            float ov  = sigm(go);
            float zv  = tanhf(gz);
            float ibv = sigm(gib);
            float fbv = sigm(gfb);
            float dec = softplus1(gd);         // BETA = 1
            float dtv = dtl[mhalf * 8 + r];
            float ci  = fv * c[r] + iv * zv;
            float cbn = fbv * cbv[r] + ibv * zv;
            float cn  = cbn + (ci - cbn) * __expf(-dec * dtv);
            float hn  = ov * tanhf(cn);
            c[r]   = cn;
            cbv[r] = cbn;
            hrow[(size_t)r * HID] = hn;                                  // h_out
            Alds[(mhalf * 8 + r) * KDIM + 2 * HID + hc + nloc] = f2bf(hn); // next A
        }
        __syncthreads();   // h / dtl writes visible before next phase 1
    }
}

// ---------------------------------------------------------------------------
// Kernel 2: h_final[b, :] = h_out[seq_len[b]-1, 0, :]
// ---------------------------------------------------------------------------
__global__ void final_kernel(const int* __restrict__ seq,
                             const float* __restrict__ h_out,
                             float* __restrict__ h_final) {
    int b = blockIdx.x;
    int j = threadIdx.x;
    int tsel = seq[b] - 1;
    h_final[b * HID + j] = h_out[(size_t)tsel * (B_TOT * HID) + j];
}

extern "C" void kernel_launch(void* const* d_in, const int* in_sizes, int n_in,
                              void* d_out, int out_size, void* d_ws, size_t ws_size,
                              hipStream_t stream) {
    const int*   items = (const int*)d_in[0];
    const float* tint  = (const float*)d_in[1];
    const int*   durs  = (const int*)d_in[2];
    const int*   seq   = (const int*)d_in[3];
    const float* iemb  = (const float*)d_in[4];
    const float* demb  = (const float*)d_in[5];
    const float* W     = (const float*)d_in[6];
    const float* bias  = (const float*)d_in[7];

    float* h_out   = (float*)d_out;
    float* h_final = h_out + (size_t)T_STEPS * B_TOT * HID;
    unsigned short* Wbf = (unsigned short*)d_ws;   // 384*896 bf16 = 688 KB

    wconv_kernel<<<(KDIM * GDIM + 255) / 256, 256, 0, stream>>>(W, Wbf);
    ctlstm_kernel<<<B_TOT / BT, 256, 0, stream>>>(items, tint, durs, iemb, demb,
                                                  bias, Wbf, h_out);
    final_kernel<<<B_TOT, HID, 0, stream>>>(seq, h_out, h_final);
}
